// GraphAttentionLayer_35794257444915
// MI455X (gfx1250) — matrix operand order repaired and checked by reading, not compiled, not run on hardware
//
#include <hip/hip_runtime.h>
#include <hip/hip_bf16.h>

#define NN 4096
#define DD 4096
#define EPSC 1e-7f
#define NEGC -1e9f

typedef __bf16 bf16_t;
typedef __attribute__((ext_vector_type(16))) __bf16 v16bf;
typedef __attribute__((ext_vector_type(8)))  __bf16 v8bf;
typedef __attribute__((ext_vector_type(8)))  float  v8f;

#define WMMA_BF16(a, b, c) \
  __builtin_amdgcn_wmma_f32_16x16x32_bf16(false, (a), false, (b), (short)0, (c), false, false)

// A fragment: 16x32 bf16 tile, rows row0..row0+15, cols k0..k0+31, row-major ld.
// Lane m=lane&15 holds K = kb..kb+7 (VGPR0-3) and 16+kb..16+kb+7 (VGPR4-7),
// kb = 8*(lane>>4).
static __device__ __forceinline__ v16bf frag_a(const bf16_t* __restrict__ base,
                                               int ld, int row0, int k0, int lane) {
  const int m  = lane & 15;
  const int kb = (lane >> 4) * 8;
  const bf16_t* p = base + (size_t)(row0 + m) * ld + (k0 + kb);
  v8bf lo = *(const v8bf*)(p);
  v8bf hi = *(const v8bf*)(p + 16);
  return __builtin_shufflevector(lo, hi, 0,1,2,3,4,5,6,7,8,9,10,11,12,13,14,15);
}

// B fragment: 32x16 bf16 tile (KxN). Lane l holds column N=col0+(l&15),
// K = kb..kb+7 and 16+kb..16+kb+7, kb = 8*(l>>4), from a row-major KxN source.
static __device__ __forceinline__ v16bf frag_b(const bf16_t* __restrict__ base,
                                               int ld, int k0, int col0, int lane) {
  const int kb = (lane >> 4) * 8;
  const bf16_t* p = base + (size_t)(k0 + kb) * ld + col0 + (lane & 15);
  v16bf r;
#pragma unroll
  for (int s = 0; s < 8; ++s) {
    r[s] = p[(size_t)s * ld];
    r[s + 8] = p[(size_t)(s + 16) * ld];
  }
  return r;
}

// ---------------------------------------------------------------- column norms
// 64 blocks x 256 threads; each block owns 64 columns, 4 row-segments/column.
__global__ __launch_bounds__(256) void gat_colnorm(const float* __restrict__ x,
                                                   float* __restrict__ cn) {
  __shared__ float red[256];
  const int dl  = threadIdx.x & 63;
  const int seg = threadIdx.x >> 6;                 // 0..3
  const int d   = blockIdx.x * 64 + dl;
  float s = 0.0f;
  const int n0 = seg * (NN / 4);
  for (int n = n0; n < n0 + NN / 4; ++n) {
    float v = x[(size_t)n * DD + d];
    s = fmaf(v, v, s);
  }
  red[threadIdx.x] = s;
  __syncthreads();
  if (seg == 0) cn[d] = sqrtf(red[dl] + red[dl + 64] + red[dl + 128] + red[dl + 192]);
}

// ------------------------------------------- f32 -> bf16 convert + transpose
__global__ __launch_bounds__(256) void gat_cvt(const float* __restrict__ x,
                                               bf16_t* __restrict__ Xb,
                                               bf16_t* __restrict__ XT) {
  __shared__ bf16_t tile[64][65];
  const int tx = threadIdx.x & 63;
  const int ty = threadIdx.x >> 6;           // 0..3
  const int c0 = blockIdx.x * 64;
  const int r0 = blockIdx.y * 64;
#pragma unroll
  for (int rr = 0; rr < 64; rr += 4) {
    const int row = r0 + rr + ty;
    const float v = x[(size_t)row * DD + c0 + tx];
    const bf16_t b = (bf16_t)v;
    Xb[(size_t)row * DD + c0 + tx] = b;
    tile[rr + ty][tx] = b;
  }
  __syncthreads();
#pragma unroll
  for (int rr = 0; rr < 64; rr += 4) {
    const int col = c0 + rr + ty;
    XT[(size_t)col * NN + r0 + tx] = tile[tx][rr + ty];
  }
}

// ------------------- GEMM1: S = beta * (x x^T) / (c_i c_j + eps) + mask ------
// One wave computes a 32x64 tile of S (2x4 grid of 16x16 WMMA accumulators).
__global__ __launch_bounds__(128) void gat_scores(const bf16_t* __restrict__ Xb,
                                                  const bf16_t* __restrict__ XT,
                                                  const float* __restrict__ cn,
                                                  const float* __restrict__ adj,
                                                  const float* __restrict__ beta,
                                                  float* __restrict__ S) {
  const int lane = threadIdx.x & 31;
  const int wid  = blockIdx.x * 4 + (threadIdx.x >> 5);
  const int TJ = NN / 64;                  // 64
  const int i0 = (wid / TJ) * 32;
  const int j0 = (wid % TJ) * 64;

  v8f acc[2][4] = {};
  for (int k0 = 0; k0 < DD; k0 += 32) {
    // Unconditional speculative prefetch (overrun stays inside the workspace).
    __builtin_prefetch(Xb + (size_t)(i0 + lane) * DD + k0 + 128, 0, 3);
    __builtin_prefetch(XT + (size_t)(k0 + 128 + lane) * NN + j0, 0, 3);
    v16bf a0 = frag_a(Xb, DD, i0,      k0, lane);
    v16bf a1 = frag_a(Xb, DD, i0 + 16, k0, lane);
    v16bf b[4];
#pragma unroll
    for (int t = 0; t < 4; ++t) b[t] = frag_b(XT, NN, k0, j0 + 16 * t, lane);
#pragma unroll
    for (int t = 0; t < 4; ++t) {
      acc[0][t] = WMMA_BF16(a0, b[t], acc[0][t]);
      acc[1][t] = WMMA_BF16(a1, b[t], acc[1][t]);
    }
  }

  const float bet = beta[0];
  const int mr = 8 * (lane >> 4);
  const int nc = lane & 15;
#pragma unroll
  for (int bi = 0; bi < 2; ++bi) {
#pragma unroll
    for (int bj = 0; bj < 4; ++bj) {
      const int gj = j0 + 16 * bj + nc;
      const float cj = cn[gj];
#pragma unroll
      for (int r = 0; r < 8; ++r) {
        const int gi = i0 + 16 * bi + mr + r;
        const size_t idx = (size_t)gi * NN + gj;
        const float sc = bet * acc[bi][bj][r] / (cn[gi] * cj + EPSC);
        S[idx] = sc + (1.0f - adj[idx]) * NEGC;
      }
    }
  }
}

// --------------------------------- row softmax: S (f32) -> P (bf16, in ws) ---
__global__ __launch_bounds__(256) void gat_softmax(const float* __restrict__ S,
                                                   bf16_t* __restrict__ Pb) {
  __shared__ float red[256];
  const int row = blockIdx.x;
  const int t = threadIdx.x;
  const float* Sr = S + (size_t)row * NN;

  float vals[16];
  float mx = -3.4e38f;
#pragma unroll
  for (int k = 0; k < 16; ++k) {
    vals[k] = Sr[t + 256 * k];
    mx = fmaxf(mx, vals[k]);
  }
  red[t] = mx;
  __syncthreads();
  for (int off = 128; off > 0; off >>= 1) {
    if (t < off) red[t] = fmaxf(red[t], red[t + off]);
    __syncthreads();
  }
  mx = red[0];
  __syncthreads();

  float sum = 0.0f;
#pragma unroll
  for (int k = 0; k < 16; ++k) {
    vals[k] = expf(vals[k] - mx);
    sum += vals[k];
  }
  red[t] = sum;
  __syncthreads();
  for (int off = 128; off > 0; off >>= 1) {
    if (t < off) red[t] += red[t + off];
    __syncthreads();
  }
  const float inv = 1.0f / red[0];

  bf16_t* Pr = Pb + (size_t)row * NN;
#pragma unroll
  for (int k = 0; k < 16; ++k) Pr[t + 256 * k] = (bf16_t)(vals[k] * inv);
}

// --------------------------------------------- GEMM2: out = P (bf16) @ x -----
// One wave computes a 32x64 tile of out (2x4 grid of accumulators).
__global__ __launch_bounds__(128) void gat_out(const bf16_t* __restrict__ Pb,
                                               const bf16_t* __restrict__ Xb,
                                               float* __restrict__ out) {
  const int lane = threadIdx.x & 31;
  const int wid  = blockIdx.x * 4 + (threadIdx.x >> 5);
  const int TJ = DD / 64;                  // 64
  const int i0 = (wid / TJ) * 32;
  const int d0 = (wid % TJ) * 64;

  v8f acc[2][4] = {};
  for (int k0 = 0; k0 < NN; k0 += 32) {
    __builtin_prefetch(Pb + (size_t)(i0 + lane) * NN + k0 + 128, 0, 3);
    __builtin_prefetch(Xb + (size_t)(k0 + 128 + lane) * DD + d0, 0, 3);
    v16bf a0 = frag_a(Pb, NN, i0,      k0, lane);
    v16bf a1 = frag_a(Pb, NN, i0 + 16, k0, lane);
    v16bf b[4];
#pragma unroll
    for (int t = 0; t < 4; ++t) b[t] = frag_b(Xb, DD, k0, d0 + 16 * t, lane);
#pragma unroll
    for (int t = 0; t < 4; ++t) {
      acc[0][t] = WMMA_BF16(a0, b[t], acc[0][t]);
      acc[1][t] = WMMA_BF16(a1, b[t], acc[1][t]);
    }
  }

  const int mr = 8 * (lane >> 4);
  const int nc = lane & 15;
#pragma unroll
  for (int bi = 0; bi < 2; ++bi) {
#pragma unroll
    for (int bj = 0; bj < 4; ++bj) {
      const int gd = d0 + 16 * bj + nc;
#pragma unroll
      for (int r = 0; r < 8; ++r) {
        const int gi = i0 + 16 * bi + mr + r;
        out[(size_t)gi * DD + gd] = acc[bi][bj][r];
      }
    }
  }
}

extern "C" void kernel_launch(void* const* d_in, const int* in_sizes, int n_in,
                              void* d_out, int out_size, void* d_ws, size_t ws_size,
                              hipStream_t stream) {
  const float* x    = (const float*)d_in[0];
  const float* adj  = (const float*)d_in[1];
  const float* beta = (const float*)d_in[2];
  float* out = (float*)d_out;

  char* ws = (char*)d_ws;
  const size_t XBYTES = (size_t)NN * DD * sizeof(bf16_t);   // 32 MB
  float*  cn = (float*)ws;                                  // 16 KB (pad to 64 KB)
  bf16_t* Xb = (bf16_t*)(ws + (1 << 16));                   // 32 MB, x in bf16
  bf16_t* XT = (bf16_t*)(ws + (1 << 16) + XBYTES);          // 32 MB, x^T in bf16
  float*  S  = (float*)(ws + (1 << 16) + 2 * XBYTES);       // 64 MB, scores f32
  bf16_t* Pb = XT;  // P reuses the x^T region (x^T dead after gat_scores)

  gat_colnorm<<<DD / 64, 256, 0, stream>>>(x, cn);
  gat_cvt<<<dim3(DD / 64, NN / 64), 256, 0, stream>>>(x, Xb, XT);
  gat_scores<<<(NN / 32) * (NN / 64) / 4, 128, 0, stream>>>(Xb, XT, cn, adj, beta, S);
  gat_softmax<<<NN, 256, 0, stream>>>(S, Pb);
  gat_out<<<(NN / 32) * (DD / 64) / 4, 128, 0, stream>>>(Pb, Xb, out);
}
